// PerLangEmbedding_22479858827436
// MI455X (gfx1250) — matrix operand order repaired and checked by reading, compiled, not run on hardware
//
#include <hip/hip_runtime.h>
#include <cstdint>

typedef __attribute__((ext_vector_type(16))) __bf16 v16bf;
typedef __attribute__((ext_vector_type(8)))  float  v8f;

#define DDIM  1024   // embedding / model dim (K and E)
#define PDIM  2048   // tokens per sequence
#define NSEQ  32     // sequences

#define TILE_P 128
#define TILE_E 128
#define TILE_K 32
#define LDS_ROW_U16 40   // 32 bf16 + 8 pad -> 80B row stride (5 uint4), conflict-free

// Single-instruction packed fp32->bf16 (RNE): result.lo = bf16(a), result.hi = bf16(b)
__device__ __forceinline__ uint32_t pk_bf16(float a, float b) {
  uint32_t r;
  asm volatile("v_cvt_pk_bf16_f32 %0, %1, %2" : "=v"(r) : "v"(a), "v"(b));
  return r;
}

union Frag { uint4 q[2]; v16bf v; };
union AccU { v8f  v;    float f[8]; };

__global__ __launch_bounds__(256)
void plembed_gemm(const int*   __restrict__ seq,
                  const float* __restrict__ embed,
                  const float* __restrict__ W,
                  const float* __restrict__ bias,
                  float*       __restrict__ out)
{
  __shared__ __align__(16) uint16_t ldsA[TILE_P * LDS_ROW_U16];
  __shared__ __align__(16) uint16_t ldsB[TILE_E * LDS_ROW_U16];

  const int tid = threadIdx.x;
  const int et  = blockIdx.x;            // E tile
  const int pt  = blockIdx.y;            // P tile
  const int n   = blockIdx.z;            // sequence
  const int p0  = pt * TILE_P;
  const int e0  = et * TILE_E;

  const int lang = seq[n * PDIM];        // language id = first token
  const float* Wl = W    + (size_t)lang * DDIM * DDIM;
  const float* bl = bias + (size_t)lang * DDIM;

  // ---- staging assignment: 2 threads per tile row, 16 floats each ----
  const int srow = tid >> 1;             // 0..127
  const int scol = (tid & 1) * 16;       // float offset within row
  const int tok  = seq[n * PDIM + p0 + srow];
  const float* gA = embed + (size_t)tok * DDIM + scol;
  const float* gB = Wl + (size_t)(e0 + srow) * DDIM + scol;
  uint4* sA = (uint4*)(ldsA + srow * LDS_ROW_U16 + scol);  // byte off: srow*80 + scol*2
  uint4* sB = (uint4*)(ldsB + srow * LDS_ROW_U16 + scol);

  // ---- wave/lane decomposition: 8 waves = 4(M) x 2(N); wave tile 32x64 ----
  const int lane = tid & 31;
  const int wid  = tid >> 5;
  const int wm   = wid & 3;              // row group of 32
  const int wn   = wid >> 2;             // col group of 64
  const int l15  = lane & 15;
  const int lh   = lane >> 4;            // lane half (WMMA layout selector)

  v8f acc[2][4];
  const v8f vzero = {0.f,0.f,0.f,0.f,0.f,0.f,0.f,0.f};
  #pragma unroll
  for (int mi = 0; mi < 2; ++mi)
    #pragma unroll
    for (int ni = 0; ni < 4; ++ni) acc[mi][ni] = vzero;

  for (int k0 = 0; k0 < DDIM; k0 += TILE_K) {
    // global fp32 loads (issued before the barrier to overlap with prior compute)
    const float4* pa = (const float4*)(gA + k0);
    const float4* pb = (const float4*)(gB + k0);
    float4 a0 = pa[0], a1 = pa[1], a2 = pa[2], a3 = pa[3];
    float4 b0 = pb[0], b1 = pb[1], b2 = pb[2], b3 = pb[3];
    if (k0 + TILE_K < DDIM) {
      __builtin_prefetch(gA + k0 + TILE_K, 0, 3);   // near-scope prefetch of next K slice
      __builtin_prefetch(gB + k0 + TILE_K, 0, 3);
    }

    __syncthreads();   // previous iteration done reading LDS
    sA[0] = make_uint4(pk_bf16(a0.x,a0.y), pk_bf16(a0.z,a0.w),
                       pk_bf16(a1.x,a1.y), pk_bf16(a1.z,a1.w));
    sA[1] = make_uint4(pk_bf16(a2.x,a2.y), pk_bf16(a2.z,a2.w),
                       pk_bf16(a3.x,a3.y), pk_bf16(a3.z,a3.w));
    sB[0] = make_uint4(pk_bf16(b0.x,b0.y), pk_bf16(b0.z,b0.w),
                       pk_bf16(b1.x,b1.y), pk_bf16(b1.z,b1.w));
    sB[1] = make_uint4(pk_bf16(b2.x,b2.y), pk_bf16(b2.z,b2.w),
                       pk_bf16(b3.x,b3.y), pk_bf16(b3.z,b3.w));
    __syncthreads();   // tiles visible

    // ---- fragment loads straight into documented WMMA VGPR layout ----
    Frag afr[2], bfr[4];
    #pragma unroll
    for (int mi = 0; mi < 2; ++mi) {
      const int row = wm * 32 + mi * 16 + l15;            // A row M = lane&15
      // lane<16: K0-7 then K16-23 ; lane>=16: K8-15 then K24-31
      afr[mi].q[0] = ((const uint4*)ldsA)[row * 5 + lh];
      afr[mi].q[1] = ((const uint4*)ldsA)[row * 5 + lh + 2];
    }
    #pragma unroll
    for (int ni = 0; ni < 4; ++ni) {
      const int row = wn * 64 + ni * 16 + l15;            // B col N = W row e
      // lane<16: K0-7 then K16-23 ; lane>=16: K8-15 then K24-31
      bfr[ni].q[0] = ((const uint4*)ldsB)[row * 5 + lh];
      bfr[ni].q[1] = ((const uint4*)ldsB)[row * 5 + lh + 2];
    }

    #pragma unroll
    for (int mi = 0; mi < 2; ++mi)
      #pragma unroll
      for (int ni = 0; ni < 4; ++ni)
        acc[mi][ni] = __builtin_amdgcn_wmma_f32_16x16x32_bf16(
            false, afr[mi].v, false, bfr[ni].v,
            (short)0, acc[mi][ni], false, false);
  }

  // ---- epilogue: bias add + store per C/D layout (M = vgpr + 8*lanehalf, N = lane&15) ----
  float bvals[4];
  #pragma unroll
  for (int ni = 0; ni < 4; ++ni)
    bvals[ni] = bl[e0 + wn * 64 + ni * 16 + l15];

  #pragma unroll
  for (int mi = 0; mi < 2; ++mi) {
    #pragma unroll
    for (int ni = 0; ni < 4; ++ni) {
      AccU u; u.v = acc[mi][ni];
      const int e = e0 + wn * 64 + ni * 16 + l15;
      #pragma unroll
      for (int v = 0; v < 8; ++v) {
        const int p = p0 + wm * 32 + mi * 16 + lh * 8 + v;
        out[((size_t)n * PDIM + p) * DDIM + e] = u.f[v] + bvals[ni];
      }
    }
  }
}

extern "C" void kernel_launch(void* const* d_in, const int* in_sizes, int n_in,
                              void* d_out, int out_size, void* d_ws, size_t ws_size,
                              hipStream_t stream) {
  (void)in_sizes; (void)n_in; (void)out_size; (void)d_ws; (void)ws_size;
  const int*   seq   = (const int*)d_in[0];
  const float* embed = (const float*)d_in[1];
  const float* W     = (const float*)d_in[2];
  const float* b     = (const float*)d_in[3];
  float*       out   = (float*)d_out;

  dim3 grid(DDIM / TILE_E, PDIM / TILE_P, NSEQ);   // 8 x 16 x 32
  dim3 block(256);                                 // 8 waves (wave32)
  hipLaunchKernelGGL(plembed_gemm, grid, block, 0, stream, seq, embed, W, b, out);
}
